// SLGAttention_23441931501667
// MI455X (gfx1250) — matrix operand order repaired and checked by reading, not compiled, not run on hardware
//
#include <hip/hip_runtime.h>

// ---------------------------------------------------------------------------
// SLG attention for MI455X (gfx1250): bf16 WMMA implicit-GEMM convs with
// async global->LDS double-buffered pipelines + flash WMMA attention with
// LDS-shared K/V. wave32; A/B fragments use contiguous k = 16*(lane/16) + s.
// ---------------------------------------------------------------------------

typedef __attribute__((ext_vector_type(16))) __bf16 v16bf;
typedef __attribute__((ext_vector_type(8)))  __bf16 v8bf;
typedef __attribute__((ext_vector_type(8)))  float  v8f;
typedef __attribute__((ext_vector_type(4)))  int    v4i;

#define Bb 2
#define Ll 4
#define Cc 128
#define VOX 4096          // 16*16*16
#define NQ  16384         // L*VOX
#define HEADS 8
#define HD 16
#define NK 1032           // 8 coarse + 1024 fine tokens
#define NKP 1056          // padded to multiple of 32
#define KTAPS 27
#define ATT_SCALE 0.25f   // HD^-0.5

#if defined(__gfx1250__) && __has_builtin(__builtin_amdgcn_global_load_async_to_lds_b128)
#define HAS_ASYNC_LDS 1
#else
#define HAS_ASYNC_LDS 0
#endif

#if HAS_ASYNC_LDS
typedef __attribute__((address_space(1))) v4i gv4i;   // global 16B vector
typedef __attribute__((address_space(3))) v4i lv4i;   // LDS 16B vector
#endif

__device__ __forceinline__ void async_g2l_b128(const unsigned short* g, unsigned short* l) {
#if HAS_ASYNC_LDS
  __builtin_amdgcn_global_load_async_to_lds_b128((gv4i*)g, (lv4i*)l, 0, 0);
#else
  *(v8bf*)l = *(const v8bf*)g;   // sync fallback: global load + ds_store
#endif
}

__device__ __forceinline__ void wait_async_le1() {
#if HAS_ASYNC_LDS
#if __has_builtin(__builtin_amdgcn_s_wait_asynccnt)
  __builtin_amdgcn_s_wait_asynccnt(1);
#else
  asm volatile("s_wait_asynccnt 0x1" ::: "memory");
#endif
#endif
}

__device__ __forceinline__ void wait_async_le0() {
#if HAS_ASYNC_LDS
#if __has_builtin(__builtin_amdgcn_s_wait_asynccnt)
  __builtin_amdgcn_s_wait_asynccnt(0);
#else
  asm volatile("s_wait_asynccnt 0x0" ::: "memory");
#endif
#endif
}

// Raw workgroup split-barrier: no vmem fence, so global prefetches stay in
// flight across it (async->LDS visibility is handled by s_wait_asynccnt, and
// ds-load results are dscnt-waited before their consuming WMMA).
__device__ __forceinline__ void block_barrier() {
#if defined(__gfx1250__) && __has_builtin(__builtin_amdgcn_s_barrier_signal) && \
    __has_builtin(__builtin_amdgcn_s_barrier_wait)
  __builtin_amdgcn_s_barrier_signal(-1);
  __builtin_amdgcn_s_barrier_wait(-1);
#else
  __syncthreads();
#endif
}

__device__ __forceinline__ unsigned short f2bs(float x) {
  union { float f; unsigned int u; } v; v.f = x;
  unsigned int u = v.u;
  u += 0x7fffu + ((u >> 16) & 1u);   // round to nearest even
  return (unsigned short)(u >> 16);
}

// ---------------- pre/post elementwise kernels -----------------------------

__global__ __launch_bounds__(128)
void zero_pad_k(unsigned short* __restrict__ z) { z[threadIdx.x] = 0; }

// x [s][c][vox] f32 -> xt [s][vox][c] bf16   (s = b*L+l, vox=4096)
__global__ __launch_bounds__(256)
void chlast_cast_k(const float* __restrict__ x, unsigned short* __restrict__ xt) {
  int i = blockIdx.x * 256 + threadIdx.x;            // over 8*4096*128
  int c = i & 127; int t = i >> 7;
  int v = t & 4095; int s = t >> 12;
  xt[i] = f2bs(x[((long)s * Cc + c) * VOX + v]);
}

// w [co][ci][3][3][3] f32 -> wt [co][tap][ci] bf16
__global__ __launch_bounds__(256)
void prep_w_k(const float* __restrict__ w, unsigned short* __restrict__ wt) {
  int i = blockIdx.x * 256 + threadIdx.x;            // over 128*128*27
  if (i >= Cc * Cc * KTAPS) return;
  int co = i / (Cc * KTAPS); int rr = i - co * (Cc * KTAPS);
  int ci = rr / KTAPS; int tap = rr - ci * KTAPS;
  wt[((long)co * KTAPS + tap) * Cc + ci] = f2bs(w[i]);
}

__global__ __launch_bounds__(256)
void cast_bf16_k(const float* __restrict__ a, unsigned short* __restrict__ o, int n) {
  int i = blockIdx.x * 256 + threadIdx.x;
  if (i < n) o[i] = f2bs(a[i]);
}

// y_q [b][nq][c] f32 -> qh [b][h][nq][32] bf16 (scaled, HD padded to 32)
__global__ __launch_bounds__(256)
void qprep_k(const float* __restrict__ yq, unsigned short* __restrict__ qh) {
  int i = blockIdx.x * 256 + threadIdx.x;            // over 2*8*16384*16
  int hd = i & 15; int t = i >> 4;
  int nq = t & (NQ - 1); int t2 = t >> 14;
  int h = t2 & 7; int b = t2 >> 3;
  float v = yq[((long)b * NQ + nq) * Cc + h * HD + hd] * ATT_SCALE;
  long o = ((long)(b * HEADS + h) * NQ + nq) * 32;
  qh[o + hd] = f2bs(v);
  qh[o + 16 + hd] = 0;
}

// y_out [s][vox][c] f32 -> out [s][c][vox] f32
__global__ __launch_bounds__(256)
void out_tr_k(const float* __restrict__ yo, float* __restrict__ out) {
  int i = blockIdx.x * 256 + threadIdx.x;            // over 8*128*4096 (out order)
  int vox = i & 4095; int t = i >> 12;
  int c = t & 127; int s = t >> 7;
  out[i] = yo[((long)s * VOX + vox) * Cc + c];
}

// ---------------- implicit-GEMM conv3d (3x3x3, pad 1) ----------------------
// in:  [slice][vox_in][128] bf16 channel-last
// wt:  [co][27][128] bf16
// out: [slice][vox_out][128] f32
// block = 256 (8 waves); wave: 16 voxels x 16 channels, K = 27*128.
// A-tile double-buffered in LDS via async global->LDS; B prefetched 1 tap
// ahead with ping-ponged register buffers (taps manually unrolled by 2).
__global__ __launch_bounds__(256)
void conv3d_wmma_k(const unsigned short* __restrict__ inp,
                   const unsigned short* __restrict__ wt,
                   const float* __restrict__ bias,
                   float* __restrict__ outp,
                   const unsigned short* __restrict__ zpad,
                   int nl, int Di, int Hi, int Wi, int Do, int Ho, int Wo,
                   int sD, int sH, int sW,
                   long in_sb, long in_sl, long out_sb, long out_sl) {
  __shared__ __align__(16) unsigned short Ash[2][16 * 136];  // row stride 136 (bank spread)
  int s = blockIdx.y; int b = s / nl; int lo = s - b * nl;
  const unsigned short* in_s = inp + b * in_sb + lo * in_sl;
  float* out_s = outp + b * out_sb + lo * out_sl;
  int nvox = Do * Ho * Wo;
  int m0 = blockIdx.x * 16;
  int tid = threadIdx.x;
  int lane = tid & 31, wave = tid >> 5;
  int n = lane & 15, hi = lane >> 4;
  int co = wave * 16 + n;
  int kbA = hi * 16, kbB = hi * 16;

  // cooperative A loader: thread fills (row = tid>>4, 8-elem chunk = tid&15)
  int lrow = tid >> 4, lchunk = tid & 15;
  int pl = m0 + lrow;
  int odl = pl / (Ho * Wo); int rr0 = pl - odl * (Ho * Wo);
  int ohl = rr0 / Wo; int owl = rr0 - ohl * Wo;
  bool prowl = (pl < nvox);
  int id0 = odl * sD - 1, ih0 = ohl * sH - 1, iw0 = owl * sW - 1;
  unsigned short* lds_dst = &Ash[0][lrow * 136 + lchunk * 8];

  auto issueA = [&](int tap, int buf) {
    int kd = tap / 9; int r2 = tap - kd * 9;
    int kh = r2 / 3;  int kw = r2 - kh * 3;
    int id = id0 + kd, ih = ih0 + kh, iw = iw0 + kw;
    bool ok = prowl && (unsigned)id < (unsigned)Di && (unsigned)ih < (unsigned)Hi &&
              (unsigned)iw < (unsigned)Wi;
    const unsigned short* g =
        ok ? in_s + (((long)id * Hi + ih) * Wi + iw) * Cc + lchunk * 8 : zpad;
    async_g2l_b128(g, lds_dst + buf * (16 * 136));
  };

  auto loadB = [&](int tap, v16bf* dst) {
    const unsigned short* brow = wt + ((long)co * KTAPS + tap) * Cc + kbB;
#pragma unroll
    for (int g = 0; g < 4; ++g) {
      v8bf b0 = ((const v8bf*)(brow + g * 32))[0];
      v8bf b1 = ((const v8bf*)(brow + g * 32))[1];
#pragma unroll
      for (int j = 0; j < 8; ++j) { dst[g][j] = b0[j]; dst[g][8 + j] = b1[j]; }
    }
  };

  v16bf B0[4], B1[4];
  issueA(0, 0);
  loadB(0, B0);
  v8f acc = {};
  const unsigned short* arow = &Ash[0][n * 136];

  // one pipeline stage: compute tap t from LDS buf (t parity), prefetch t+1
  auto step = [&](int t, int par, v16bf* Buse, v16bf* Bload) {
    block_barrier();                         // buffer (t+1)&1 free everywhere
    if (t + 1 < KTAPS) {
      issueA(t + 1, par ^ 1);
      loadB(t + 1, Bload);
      wait_async_le1();                      // tap t landed; tap t+1 in flight
    } else {
      wait_async_le0();
    }
    block_barrier();                         // tap t visible to all waves
    const unsigned short* ab = arow + par * (16 * 136);
    v8bf a0[4], a1[4];
#pragma unroll
    for (int g = 0; g < 4; ++g) {            // stage all A-frag loads first
      a0[g] = *(const v8bf*)(ab + g * 32 + kbA);
      a1[g] = *(const v8bf*)(ab + g * 32 + kbA + 8);
    }
#pragma unroll
    for (int g = 0; g < 4; ++g) {            // then 4 back-to-back WMMAs
      v16bf a;
#pragma unroll
      for (int j = 0; j < 8; ++j) { a[j] = a0[g][j]; a[8 + j] = a1[g][j]; }
      acc = __builtin_amdgcn_wmma_f32_16x16x32_bf16(false, a, false, Buse[g], (short)0,
                                                    acc, false, false);
    }
  };

  for (int t = 0; t < KTAPS; t += 2) {       // manual 2x unroll: no B reg copies
    step(t, 0, B0, B1);
    if (t + 1 < KTAPS) step(t + 1, 1, B1, B0);
  }

  float bvv = bias[co];
#pragma unroll
  for (int r = 0; r < 8; ++r) {
    int pm = m0 + r + 8 * hi;                // D layout: M = r + 8*(lane>>4)
    if (pm < nvox) out_s[(long)pm * Cc + co] = acc[r] + bvv;
  }
}

// ---------------- LayerNorm + head split for K/V ---------------------------
// kcat/vcat [b][1032][128] f32 -> kh [b][h][1056][32] bf16 (HD+key padded),
//                                 vt [b][h][16][1056] bf16 (transposed)
__global__ __launch_bounds__(128)
void ln_kv_k(const float* __restrict__ kcat, const float* __restrict__ vcat,
             const float* __restrict__ gam, const float* __restrict__ bet,
             unsigned short* __restrict__ kh, unsigned short* __restrict__ vt) {
  int nt = blockIdx.x, b = blockIdx.y, c = threadIdx.x;
  int h = c >> 4, hd = c & 15;
  long kho = ((long)(b * HEADS + h) * NKP + nt) * 32;
  long vto = ((long)(b * HEADS + h) * HD + hd) * NKP + nt;
  if (nt >= NK) { kh[kho + hd] = 0; kh[kho + 16 + hd] = 0; vt[vto] = 0; return; }
  __shared__ float sb[128];
  float kv = kcat[((long)b * NK + nt) * Cc + c];
  float vv = vcat[((long)b * NK + nt) * Cc + c];
  float g = gam[c], bt = bet[c];
  sb[c] = kv; __syncthreads();
  for (int o = 64; o > 0; o >>= 1) { if (c < o) sb[c] += sb[c + o]; __syncthreads(); }
  float m = sb[0] * (1.f / 128.f); __syncthreads();
  float d = kv - m; sb[c] = d * d; __syncthreads();
  for (int o = 64; o > 0; o >>= 1) { if (c < o) sb[c] += sb[c + o]; __syncthreads(); }
  float kn = d * rsqrtf(sb[0] * (1.f / 128.f) + 1e-5f) * g + bt; __syncthreads();
  sb[c] = vv; __syncthreads();
  for (int o = 64; o > 0; o >>= 1) { if (c < o) sb[c] += sb[c + o]; __syncthreads(); }
  m = sb[0] * (1.f / 128.f); __syncthreads();
  d = vv - m; sb[c] = d * d; __syncthreads();
  for (int o = 64; o > 0; o >>= 1) { if (c < o) sb[c] += sb[c + o]; __syncthreads(); }
  float vn = d * rsqrtf(sb[0] * (1.f / 128.f) + 1e-5f) * g + bt;
  kh[kho + hd] = f2bs(kn);
  kh[kho + 16 + hd] = 0;
  vt[vto] = f2bs(vn);
}

// ---------------- flash attention: 8 waves/block share K/V via LDS ---------
// Each wave owns one 16-row q-tile; K/V chunks (32 keys) double-buffered in
// LDS via async global->LDS.
__global__ __launch_bounds__(256)
void attn_k(const unsigned short* __restrict__ qh,
            const unsigned short* __restrict__ kh,
            const unsigned short* __restrict__ vt,
            unsigned short* __restrict__ ot) {
  __shared__ __align__(16) unsigned short Ksh[2][32 * 32];
  __shared__ __align__(16) unsigned short Vsh[2][16 * 32];
  __shared__ __align__(16) unsigned short psh[8][16 * 32];
  int h = blockIdx.y, b = blockIdx.z;
  int tid = threadIdx.x;
  int lane = tid & 31, wave = tid >> 5;
  int n = lane & 15, hi = lane >> 4;
  int kbA = hi * 16;
  int qt = blockIdx.x * 8 + wave;                    // q-tile of this wave
  const unsigned short* qb = qh + ((long)(b * HEADS + h) * NQ + qt * 16) * 32;
  const unsigned short* kb = kh + (long)(b * HEADS + h) * NKP * 32;
  const unsigned short* vb = vt + (long)(b * HEADS + h) * HD * NKP;

  auto issueKV = [&](int c0, int buf) {
    int k0 = c0 * 32;
    if (tid < 128) {                                 // waves 0-3: K tile 32x32
      int row = tid >> 2, ch = tid & 3;
      async_g2l_b128(kb + (long)(k0 + row) * 32 + ch * 8, &Ksh[buf][row * 32 + ch * 8]);
    } else if (tid < 192) {                          // waves 4-5: V tile 16x32 (hd-major)
      int u = tid - 128; int vhd = u >> 2, ch = u & 3;
      async_g2l_b128(vb + (long)vhd * NKP + k0 + ch * 8, &Vsh[buf][vhd * 32 + ch * 8]);
    }
  };

  // Q fragment (A 16x32), row = lane&15
  v16bf qa;
  {
    v8bf t0 = *(const v8bf*)(qb + n * 32 + kbA);
    v8bf t1 = *(const v8bf*)(qb + n * 32 + kbA + 8);
#pragma unroll
    for (int j = 0; j < 8; ++j) { qa[j] = t0[j]; qa[8 + j] = t1[j]; }
  }
  float runmax[8], runsum[8];
  v8f oacc = {};
#pragma unroll
  for (int r = 0; r < 8; ++r) { runmax[r] = -1e30f; runsum[r] = 0.f; }

  const int NCH = NKP / 32;                          // 33 chunks
  issueKV(0, 0);
  for (int cch = 0; cch < NCH; ++cch) {
    block_barrier();                                 // buffer (cch+1)&1 free everywhere
    if (cch + 1 < NCH) {
      issueKV(cch + 1, (cch + 1) & 1);
      wait_async_le1();
    } else {
      wait_async_le0();
    }
    block_barrier();                                 // chunk cch in LDS
    int buf = cch & 1, k0 = cch * 32;
    const unsigned short* kp = &Ksh[buf][0];
    const unsigned short* vp = &Vsh[buf][0];
    v16bf kf0, kf1;
    {
      v8bf a0 = *(const v8bf*)(kp + n * 32 + hi * 16);
      v8bf a1 = *(const v8bf*)(kp + n * 32 + hi * 16 + 8);
      v8bf c0 = *(const v8bf*)(kp + (16 + n) * 32 + hi * 16);
      v8bf c1 = *(const v8bf*)(kp + (16 + n) * 32 + hi * 16 + 8);
#pragma unroll
      for (int j = 0; j < 8; ++j) {
        kf0[j] = a0[j]; kf0[8 + j] = a1[j];
        kf1[j] = c0[j]; kf1[8 + j] = c1[j];
      }
    }
    v8f z = {};
    v8f s0 = __builtin_amdgcn_wmma_f32_16x16x32_bf16(false, qa, false, kf0, (short)0, z, false, false);
    v8f s1 = __builtin_amdgcn_wmma_f32_16x16x32_bf16(false, qa, false, kf1, (short)0, z, false, false);
    bool v0 = (k0 + n) < NK, v1 = (k0 + 16 + n) < NK;
    unsigned short* pw = &psh[wave][0];
#pragma unroll
    for (int r = 0; r < 8; ++r) {
      float x0 = v0 ? s0[r] : -1e30f;
      float x1 = v1 ? s1[r] : -1e30f;
      float t = fmaxf(x0, x1);
      t = fmaxf(t, __shfl_xor(t, 1, 32));
      t = fmaxf(t, __shfl_xor(t, 2, 32));
      t = fmaxf(t, __shfl_xor(t, 4, 32));
      t = fmaxf(t, __shfl_xor(t, 8, 32));            // row max within 16-lane half
      float nm = fmaxf(runmax[r], t);
      float sc = __expf(runmax[r] - nm);
      float e0 = v0 ? __expf(s0[r] - nm) : 0.f;
      float e1 = v1 ? __expf(s1[r] - nm) : 0.f;
      float es = e0 + e1;
      es += __shfl_xor(es, 1, 32);
      es += __shfl_xor(es, 2, 32);
      es += __shfl_xor(es, 4, 32);
      es += __shfl_xor(es, 8, 32);
      runsum[r] = runsum[r] * sc + es;
      runmax[r] = nm;
      oacc[r] *= sc;
      pw[(r + 8 * hi) * 32 + n] = f2bs(e0);          // D layout -> row-major P
      pw[(r + 8 * hi) * 32 + 16 + n] = f2bs(e1);
    }
    block_barrier();                                 // P stores -> A-frag loads
    v16bf pa, vf;
    {
      v8bf t0 = *(const v8bf*)(pw + n * 32 + kbA);
      v8bf t1 = *(const v8bf*)(pw + n * 32 + kbA + 8);
      v8bf u0 = *(const v8bf*)(vp + n * 32 + hi * 16);
      v8bf u1 = *(const v8bf*)(vp + n * 32 + hi * 16 + 8);
#pragma unroll
      for (int j = 0; j < 8; ++j) {
        pa[j] = t0[j]; pa[8 + j] = t1[j];
        vf[j] = u0[j]; vf[8 + j] = u1[j];
      }
    }
    oacc = __builtin_amdgcn_wmma_f32_16x16x32_bf16(false, pa, false, vf, (short)0, oacc, false, false);
  }
#pragma unroll
  for (int r = 0; r < 8; ++r) {
    float val = oacc[r] / runsum[r];
    int tok = qt * 16 + r + 8 * hi;
    int c = h * HD + n;
    ot[((long)b * NQ + tok) * Cc + c] = f2bs(val);
  }
}

// ---------------------------------------------------------------------------

extern "C" void kernel_launch(void* const* d_in, const int* in_sizes, int n_in,
                              void* d_out, int out_size, void* d_ws, size_t ws_size,
                              hipStream_t stream) {
  (void)in_sizes; (void)n_in; (void)out_size; (void)ws_size;
  const float* x       = (const float*)d_in[0];
  const float* wq      = (const float*)d_in[1];
  const float* bq      = (const float*)d_in[2];
  const float* wk      = (const float*)d_in[3];
  const float* bk      = (const float*)d_in[4];
  const float* wv      = (const float*)d_in[5];
  const float* bv      = (const float*)d_in[6];
  const float* wproj   = (const float*)d_in[7];
  const float* bproj   = (const float*)d_in[8];
  const float* wfine   = (const float*)d_in[9];
  const float* bfine   = (const float*)d_in[10];
  const float* wcoarse = (const float*)d_in[11];
  const float* bcoarse = (const float*)d_in[12];
  const float* gamma   = (const float*)d_in[13];
  const float* beta    = (const float*)d_in[14];
  float* out = (float*)d_out;

  char* ws = (char*)d_ws;
  size_t off = 0;
  auto A = [&](size_t bytes) { size_t o = off; off += (bytes + 255) & ~(size_t)255; return o; };

  unsigned short* XT  = (unsigned short*)(ws + A((size_t)8 * VOX * Cc * 2));    // x ch-last bf16
  unsigned short* WQT = (unsigned short*)(ws + A((size_t)Cc * KTAPS * Cc * 2));
  unsigned short* WKT = (unsigned short*)(ws + A((size_t)Cc * KTAPS * Cc * 2));
  unsigned short* WVT = (unsigned short*)(ws + A((size_t)Cc * KTAPS * Cc * 2));
  unsigned short* WPT = (unsigned short*)(ws + A((size_t)Cc * KTAPS * Cc * 2));
  unsigned short* WFT = (unsigned short*)(ws + A((size_t)Cc * KTAPS * Cc * 2));
  unsigned short* WCT = (unsigned short*)(ws + A((size_t)Cc * KTAPS * Cc * 2));
  float* YQ  = (float*)(ws + A((size_t)Bb * NQ * Cc * 4));                      // conv_q out / y_out
  float* YF  = (float*)(ws + A((size_t)Bb * 2 * 512 * Cc * 4));
  float* YC  = (float*)(ws + A((size_t)Bb * 8 * Cc * 4));
  unsigned short* XFT = (unsigned short*)(ws + A((size_t)Bb * 2 * 512 * Cc * 2));
  unsigned short* XCT = (unsigned short*)(ws + A((size_t)Bb * 8 * Cc * 2));
  float* KCAT = (float*)(ws + A((size_t)Bb * NK * Cc * 4));
  float* VCAT = (float*)(ws + A((size_t)Bb * NK * Cc * 4));
  unsigned short* QH = (unsigned short*)(ws + A((size_t)Bb * HEADS * NQ * 32 * 2));
  unsigned short* KH = (unsigned short*)(ws + A((size_t)Bb * HEADS * NKP * 32 * 2));
  unsigned short* VT = (unsigned short*)(ws + A((size_t)Bb * HEADS * HD * NKP * 2));
  unsigned short* ZP = (unsigned short*)(ws + A((size_t)256));                  // zero page
  unsigned short* OT = XT;   // reuse: xt consumed by q/fine/coarse convs before attention
  float* YO = YQ;            // reuse: y_q consumed by qprep before proj conv

  const long SL = (long)VOX * Cc;   // per (b,l) slice elements

  zero_pad_k<<<1, 128, 0, stream>>>(ZP);
  prep_w_k<<<1728, 256, 0, stream>>>(wq, WQT);
  prep_w_k<<<1728, 256, 0, stream>>>(wk, WKT);
  prep_w_k<<<1728, 256, 0, stream>>>(wv, WVT);
  prep_w_k<<<1728, 256, 0, stream>>>(wproj, WPT);
  prep_w_k<<<1728, 256, 0, stream>>>(wfine, WFT);
  prep_w_k<<<1728, 256, 0, stream>>>(wcoarse, WCT);
  chlast_cast_k<<<16384, 256, 0, stream>>>(x, XT);

  // q = conv(x, wq, stride 1)
  conv3d_wmma_k<<<dim3(256, 8), 256, 0, stream>>>(XT, WQT, bq, YQ, ZP,
      4, 16, 16, 16, 16, 16, 16, 1, 1, 1, 4 * SL, SL, 4 * SL, SL);
  // xc = conv(x[::4], wcoarse, stride 8) -> 2x2x2
  conv3d_wmma_k<<<dim3(1, 2), 256, 0, stream>>>(XT, WCT, bcoarse, YC, ZP,
      1, 16, 16, 16, 2, 2, 2, 8, 8, 8, 4 * SL, 4 * SL, 8L * Cc, 0);
  // xf = conv(x[::2], wfine, stride 2) -> 8x8x8
  conv3d_wmma_k<<<dim3(32, 4), 256, 0, stream>>>(XT, WFT, bfine, YF, ZP,
      2, 16, 16, 16, 8, 8, 8, 2, 2, 2, 4 * SL, 2 * SL, 2L * 512 * Cc, 512L * Cc);
  cast_bf16_k<<<1024, 256, 0, stream>>>(YF, XFT, Bb * 2 * 512 * Cc);
  cast_bf16_k<<<8, 256, 0, stream>>>(YC, XCT, Bb * 8 * Cc);

  // k/v on coarse+fine, concatenated into [b][1032][128]
  conv3d_wmma_k<<<dim3(1, 2), 256, 0, stream>>>(XCT, WKT, bk, KCAT, ZP,
      1, 2, 2, 2, 2, 2, 2, 1, 1, 1, 8L * Cc, 0, (long)NK * Cc, 0);
  conv3d_wmma_k<<<dim3(32, 4), 256, 0, stream>>>(XFT, WKT, bk, KCAT + 8 * Cc, ZP,
      2, 8, 8, 8, 8, 8, 8, 1, 1, 1, 2L * 512 * Cc, 512L * Cc, (long)NK * Cc, 512L * Cc);
  conv3d_wmma_k<<<dim3(1, 2), 256, 0, stream>>>(XCT, WVT, bv, VCAT, ZP,
      1, 2, 2, 2, 2, 2, 2, 1, 1, 1, 8L * Cc, 0, (long)NK * Cc, 0);
  conv3d_wmma_k<<<dim3(32, 4), 256, 0, stream>>>(XFT, WVT, bv, VCAT + 8 * Cc, ZP,
      2, 8, 8, 8, 8, 8, 8, 1, 1, 1, 2L * 512 * Cc, 512L * Cc, (long)NK * Cc, 512L * Cc);

  // head split + scale (q), layernorm + head split (k, v)
  qprep_k<<<16384, 256, 0, stream>>>(YQ, QH);
  ln_kv_k<<<dim3(NKP, Bb), 128, 0, stream>>>(KCAT, VCAT, gamma, beta, KH, VT);

  // attention -> o tokens (channel-last bf16); 8 q-tiles per block share K/V
  attn_k<<<dim3(NQ / 16 / 8, HEADS, Bb), 256, 0, stream>>>(QH, KH, VT, OT);

  // proj conv + final layout transpose
  conv3d_wmma_k<<<dim3(256, 8), 256, 0, stream>>>(OT, WPT, bproj, YO, ZP,
      4, 16, 16, 16, 16, 16, 16, 1, 1, 1, 4 * SL, SL, 4 * SL, SL);
  out_tr_k<<<16384, 256, 0, stream>>>(YO, out);
}
